// PointNetSetAbstraction_3427383902464
// MI455X (gfx1250) — matrix operand order repaired and checked by reading, not compiled, not run on hardware
//
#include <hip/hip_runtime.h>

// Problem constants (match reference)
#define NPTS   4096
#define BATCH  16
#define SPTS   1024      // NPOINT
#define KSAMP  32        // NSAMPLE
#define DIN    64
#define ROWS   (BATCH*SPTS*KSAMP)   // 524288
#define K0     96        // 3+64=67 padded to 3 k-steps of 32
#define R2     0.04f     // RADIUS^2

typedef __attribute__((ext_vector_type(16))) _Float16 v16h;
typedef __attribute__((ext_vector_type(8)))  _Float16 v8h;
typedef __attribute__((ext_vector_type(8)))  float    v8f;

// ---------------------------------------------------------------------------
// Kernel 0: convert weights to f16 (pad W0 67->96 with zeros), zero BN stats
// ---------------------------------------------------------------------------
__global__ void prep_kernel(const float* __restrict__ W0, const float* __restrict__ W1,
                            const float* __restrict__ W2,
                            _Float16* __restrict__ W0h, _Float16* __restrict__ W1h,
                            _Float16* __restrict__ W2h, float* __restrict__ stats)
{
    const int total = 64*K0 + 64*64 + 128*64 + 1536;
    for (int i = blockIdx.x*blockDim.x + threadIdx.x; i < total; i += gridDim.x*blockDim.x) {
        int x = i;
        if (x < 64*K0) { int r = x / K0, c = x % K0;
            W0h[x] = (c < 67) ? (_Float16)W0[r*67 + c] : (_Float16)0.f; continue; }
        x -= 64*K0;
        if (x < 64*64)  { W1h[x] = (_Float16)W1[x]; continue; }
        x -= 64*64;
        if (x < 128*64) { W2h[x] = (_Float16)W2[x]; continue; }
        x -= 128*64;
        stats[x] = 0.f;   // sum/sumsq/scale/shift for 3 layers
    }
}

// ---------------------------------------------------------------------------
// Kernel 1: farthest point sampling. One block per batch; dist in registers,
// LDS tree argmax. Also writes new_xyz output (B,3,SPTS).
// ---------------------------------------------------------------------------
__global__ __launch_bounds__(256) void fps_kernel(const float* __restrict__ xyz,
                                                  int* __restrict__ fps_idx,
                                                  float* __restrict__ new_xyz)
{
    const int b   = blockIdx.x;
    const int tid = threadIdx.x;
    const float* X = xyz + (size_t)b * 3 * NPTS;   // x | y | z planes

    __shared__ float cc[3];
    __shared__ float s_red[256];
    __shared__ int   s_idx[256];

    float dist[16];
#pragma unroll
    for (int j = 0; j < 16; ++j) dist[j] = 1e10f;

    int far = 0;
    for (int it = 0; it < SPTS; ++it) {
        if (tid == 0) {
            fps_idx[b*SPTS + it] = far;
            cc[0] = X[far]; cc[1] = X[NPTS + far]; cc[2] = X[2*NPTS + far];
        }
        if (tid < 3) new_xyz[(size_t)b*3*SPTS + tid*SPTS + it] = X[tid*NPTS + far];
        __syncthreads();
        const float c0 = cc[0], c1 = cc[1], c2 = cc[2];
        float bmax = -1.f; int bidx = 0;
#pragma unroll
        for (int j = 0; j < 16; ++j) {
            const int p = j*256 + tid;
            const float dx = X[p] - c0, dy = X[NPTS + p] - c1, dz = X[2*NPTS + p] - c2;
            const float d = dx*dx + dy*dy + dz*dz;
            dist[j] = fminf(dist[j], d);
            if (dist[j] > bmax) { bmax = dist[j]; bidx = p; }
        }
        s_red[tid] = bmax; s_idx[tid] = bidx;
        __syncthreads();
        for (int st = 128; st > 0; st >>= 1) {
            if (tid < st && s_red[tid + st] > s_red[tid]) {
                s_red[tid] = s_red[tid + st]; s_idx[tid] = s_idx[tid + st];
            }
            __syncthreads();
        }
        far = s_idx[0];
        __syncthreads();
    }
}

// ---------------------------------------------------------------------------
// Kernel 2: ball query + gather + concat -> H0 (ROWS x K0, f16, zero-padded).
// One wave32 per centroid; ordered in-radius selection via ballot prefix.
// ---------------------------------------------------------------------------
__global__ __launch_bounds__(256) void group_kernel(const float* __restrict__ xyz,
                                                    const float* __restrict__ points,
                                                    const int* __restrict__ fps_idx,
                                                    _Float16* __restrict__ H0)
{
    const int gw   = (blockIdx.x*256 + threadIdx.x) >> 5;   // centroid id, 0..16383
    const int lane = threadIdx.x & 31;
    const int b = gw >> 10, s = gw & 1023;
    const float* X = xyz + (size_t)b * 3 * NPTS;
    const int fidx = fps_idx[b*SPTS + s];
    const float cx = X[fidx], cy = X[NPTS + fidx], cz = X[2*NPTS + fidx];

    int myidx = -1, first_idx = -1, cnt = 0;
    for (int base = 0; base < NPTS; base += 32) {
        const int i = base + lane;
        const float dx = X[i] - cx, dy = X[NPTS + i] - cy, dz = X[2*NPTS + i] - cz;
        const float d2 = dx*dx + dy*dy + dz*dz;
        const unsigned mask = __builtin_amdgcn_ballot_w32(d2 <= R2);
        const int nm = __popc(mask);
        if (first_idx < 0 && nm > 0) first_idx = base + __ffs(mask) - 1;
        if (myidx < 0 && lane >= cnt && lane < cnt + nm) {
            const int need = lane - cnt;
            unsigned m = mask;
            for (int j = 0; j < need; ++j) m &= m - 1;   // clear lowest 'need' bits
            myidx = base + __ffs(m) - 1;
        }
        cnt += nm;
        if (cnt >= KSAMP) break;        // uniform across wave (ballot-derived)
    }
    const int pidx = (myidx >= 0) ? myidx : (first_idx >= 0 ? first_idx : fidx);

    const size_t row = (size_t)gw * KSAMP + lane;           // (b*1024+s)*32 + k
    _Float16* h = H0 + row * K0;
    h[0] = (_Float16)(X[pidx] - cx);
    h[1] = (_Float16)(X[NPTS + pidx] - cy);
    h[2] = (_Float16)(X[2*NPTS + pidx] - cz);
    const float* P = points + (size_t)b * DIN * NPTS + pidx;
#pragma unroll 8
    for (int d = 0; d < DIN; ++d) h[3 + d] = (_Float16)P[(size_t)d * NPTS];
#pragma unroll
    for (int c = 3 + DIN; c < K0; ++c) h[c] = (_Float16)0.f;
}

// ---------------------------------------------------------------------------
// GEMM + bias + BN-stat accumulation. Y = A(ROWS x KD) * W^T(NOUT x KD) + b.
// Register-blocked: one wave owns 16 rows x NOUT (TILES_N accumulators), so
// the BN+ReLU fused A-operand unpack is done ONCE per k-step and amortized
// over TILES_N v_wmma_f32_16x16x32_f16 issues; A is fetched once per k-step.
// Raw (pre-BN) outputs stored f16; per-channel sum/sumsq accumulated in f32
// via LDS ds_add_f32 then one global atomic per channel per block.
// ---------------------------------------------------------------------------
template<int NOUT, int KD, bool BN>
__global__ __launch_bounds__(256) void gemm_bn_kernel(const _Float16* __restrict__ A,
                                                      const _Float16* __restrict__ W,
                                                      const float* __restrict__ bias,
                                                      const float* __restrict__ bnscale,
                                                      const float* __restrict__ bnshift,
                                                      _Float16* __restrict__ Y,
                                                      float* __restrict__ gsum,
                                                      float* __restrict__ gsq)
{
    constexpr int TILES_N = NOUT / 16;
    __shared__ float s_scale[KD], s_shift[KD];
    __shared__ float s_sum[NOUT], s_sq[NOUT];
    for (int i = threadIdx.x; i < NOUT; i += 256) { s_sum[i] = 0.f; s_sq[i] = 0.f; }
    if (BN) for (int i = threadIdx.x; i < KD; i += 256) { s_scale[i] = bnscale[i]; s_shift[i] = bnshift[i]; }
    __syncthreads();

    const int wave = threadIdx.x >> 5, lane = threadIdx.x & 31;
    const int lr = lane & 15, hs = lane >> 4;            // half-wave select
    const size_t tm = (size_t)blockIdx.x*8 + wave;       // row-tile, 0..ROWS/16-1
    const size_t rb = tm * 16;

    const _Float16* Arow = A + (rb + lr) * (size_t)KD;              // row rb+lr
    const _Float16* Wcol = W + (size_t)lr * KD + hs*8;              // col lr of tile 0

    v8f acc[TILES_N];
#pragma unroll
    for (int t = 0; t < TILES_N; ++t) acc[t] = (v8f){};

    for (int kb = 0; kb < KD; kb += 32) {
        const int off0 = kb + hs*8;
        // ISA 16-bit A 16x32 layout: lanes<16 -> K {kb..kb+7, kb+16..kb+23},
        // lanes>=16 -> K {kb+8..kb+15, kb+24..kb+31}; each half = one b128 load
        v8h a0 = *(const v8h*)(Arow + off0);
        v8h a1 = *(const v8h*)(Arow + off0 + 16);
        v16h av;
        if (BN) {
#pragma unroll
            for (int t = 0; t < 8; ++t) {
                const float y0 = (float)a0[t] * s_scale[off0 + t]      + s_shift[off0 + t];
                const float y1 = (float)a1[t] * s_scale[off0 + 16 + t] + s_shift[off0 + 16 + t];
                av[t]     = (_Float16)fmaxf(y0, 0.f);
                av[t + 8] = (_Float16)fmaxf(y1, 0.f);
            }
        } else {
#pragma unroll
            for (int t = 0; t < 8; ++t) { av[t] = a0[t]; av[t + 8] = a1[t]; }
        }
#pragma unroll
        for (int tn = 0; tn < TILES_N; ++tn) {
            const _Float16* Wp = Wcol + (size_t)tn * 16 * KD + kb;
            v8h b0 = *(const v8h*)(Wp);
            v8h b1 = *(const v8h*)(Wp + 16);
            v16h bv;
#pragma unroll
            for (int t = 0; t < 8; ++t) { bv[t] = b0[t]; bv[t + 8] = b1[t]; }
            acc[tn] = __builtin_amdgcn_wmma_f32_16x16x32_f16(false, av, false, bv,
                                                             (short)0, acc[tn], false, false);
        }
    }

    // Epilogue: bias, store raw f16, accumulate BN stats
#pragma unroll
    for (int tn = 0; tn < TILES_N; ++tn) {
        const int ncol = tn*16 + lr;
        const float bb = bias[ncol];
        float ps = 0.f, pq = 0.f;
#pragma unroll
        for (int v = 0; v < 8; ++v) {                  // C/D layout: row rb+v(+8 for hs)
            const float y = acc[tn][v] + bb;
            ps += y; pq += y*y;
            const size_t r = rb + v + hs*8;
            Y[r * NOUT + ncol] = (_Float16)y;
        }
        atomicAdd(&s_sum[ncol], ps);                    // ds_add_f32
        atomicAdd(&s_sq[ncol],  pq);
    }
    __syncthreads();
    for (int i = threadIdx.x; i < NOUT; i += 256) {     // one global atomic per channel
        atomicAdd(&gsum[i], s_sum[i]);
        atomicAdd(&gsq[i],  s_sq[i]);
    }
}

// ---------------------------------------------------------------------------
// Finalize BN: scale = g*rsqrt(var+eps); shift = beta - mu*scale
// ---------------------------------------------------------------------------
__global__ void finalize_kernel(const float* __restrict__ gsum, const float* __restrict__ gsq,
                                const float* __restrict__ g, const float* __restrict__ be,
                                float* __restrict__ scale, float* __restrict__ shift, int nch)
{
    const int i = threadIdx.x;
    if (i < nch) {
        const float invc = 1.f / (float)ROWS;
        const float mu  = gsum[i] * invc;
        const float var = gsq[i] * invc - mu*mu;
        const float rs  = rsqrtf(var + 1e-5f);
        const float sc  = g[i] * rs;
        scale[i] = sc;
        shift[i] = be[i] - mu*sc;
    }
}

// ---------------------------------------------------------------------------
// BN+ReLU on layer-3 raw output fused with max over K=32; transposed store.
// ---------------------------------------------------------------------------
__global__ __launch_bounds__(256) void maxpool_kernel(const _Float16* __restrict__ Y3,
                                                      const float* __restrict__ scale,
                                                      const float* __restrict__ shift,
                                                      float* __restrict__ outp)
{
    const int t  = blockIdx.x*256 + threadIdx.x;   // 0 .. B*S*128-1
    const int n  = t & 127;
    const int bs = t >> 7;
    const _Float16* Yp = Y3 + (size_t)bs * KSAMP * 128 + n;
    const float sc = scale[n], sh = shift[n];
    float m = 0.f;                                 // max of ReLU'd values >= 0
#pragma unroll 8
    for (int kk = 0; kk < KSAMP; ++kk) {
        float y = (float)Yp[(size_t)kk * 128] * sc + sh;
        y = fmaxf(y, 0.f);
        m = fmaxf(m, y);
    }
    const int b = bs >> 10, s = bs & 1023;
    outp[(size_t)b*128*SPTS + (size_t)n*SPTS + s] = m;
}

// ---------------------------------------------------------------------------
extern "C" void kernel_launch(void* const* d_in, const int* in_sizes, int n_in,
                              void* d_out, int out_size, void* d_ws, size_t ws_size,
                              hipStream_t stream)
{
    (void)in_sizes; (void)n_in; (void)out_size; (void)ws_size;
    const float* xyz    = (const float*)d_in[0];
    const float* points = (const float*)d_in[1];
    const float* W0 = (const float*)d_in[2];  const float* b0 = (const float*)d_in[3];
    const float* g0 = (const float*)d_in[4];  const float* be0 = (const float*)d_in[5];
    const float* W1 = (const float*)d_in[6];  const float* b1 = (const float*)d_in[7];
    const float* g1 = (const float*)d_in[8];  const float* be1 = (const float*)d_in[9];
    const float* W2 = (const float*)d_in[10]; const float* b2 = (const float*)d_in[11];
    const float* g2 = (const float*)d_in[12]; const float* be2 = (const float*)d_in[13];

    char* ws = (char*)d_ws;
    size_t off = 0;
    auto carve = [&](size_t bytes) -> char* {
        char* p = ws + off;
        off = (off + bytes + 255) & ~(size_t)255;
        return p;
    };
    _Float16* W0h  = (_Float16*)carve((size_t)64*K0*2);
    _Float16* W1h  = (_Float16*)carve((size_t)64*64*2);
    _Float16* W2h  = (_Float16*)carve((size_t)128*64*2);
    int*      fps  = (int*)     carve((size_t)BATCH*SPTS*4);
    float*    stats= (float*)   carve((size_t)1536*4);      // 3 layers x {sum,sq,scale,shift}[128]
    _Float16* H0   = (_Float16*)carve((size_t)ROWS*K0*2);
    _Float16* Y1   = (_Float16*)carve((size_t)ROWS*64*2);
    _Float16* Y2   = (_Float16*)carve((size_t)ROWS*64*2);
    _Float16* Y3   = (_Float16*)carve((size_t)ROWS*128*2);

    float* sum0 = stats +   0, *sq0 = stats + 128, *sc0 = stats + 256, *sh0 = stats + 384;
    float* sum1 = stats + 512, *sq1 = stats + 640, *sc1 = stats + 768, *sh1 = stats + 896;
    float* sum2 = stats +1024, *sq2 = stats +1152, *sc2 = stats +1280, *sh2 = stats +1408;

    float* out_xyz = (float*)d_out;                         // (B,3,SPTS)
    float* out_pts = (float*)d_out + (size_t)BATCH*3*SPTS;  // (B,128,SPTS)

    prep_kernel<<<32, 256, 0, stream>>>(W0, W1, W2, W0h, W1h, W2h, stats);
    fps_kernel<<<BATCH, 256, 0, stream>>>(xyz, fps, out_xyz);
    group_kernel<<<(BATCH*SPTS)/8, 256, 0, stream>>>(xyz, points, fps, H0);

    const int gemm_blocks = (ROWS/16)/8;    // 4096: one 16-row tile per wave

    // Layer 1: H0 (ROWS x 96) * W0h^T -> Y1 raw, stats0
    gemm_bn_kernel<64, K0, false><<<gemm_blocks, 256, 0, stream>>>(
        H0, W0h, b0, nullptr, nullptr, Y1, sum0, sq0);
    finalize_kernel<<<1, 128, 0, stream>>>(sum0, sq0, g0, be0, sc0, sh0, 64);

    // Layer 2: BN0+ReLU fused into A-load; Y2 raw, stats1
    gemm_bn_kernel<64, 64, true><<<gemm_blocks, 256, 0, stream>>>(
        Y1, W1h, b1, sc0, sh0, Y2, sum1, sq1);
    finalize_kernel<<<1, 128, 0, stream>>>(sum1, sq1, g1, be1, sc1, sh1, 64);

    // Layer 3: BN1+ReLU fused into A-load; Y3 raw, stats2
    gemm_bn_kernel<128, 64, true><<<gemm_blocks, 256, 0, stream>>>(
        Y2, W2h, b2, sc1, sh1, Y3, sum2, sq2);
    finalize_kernel<<<1, 128, 0, stream>>>(sum2, sq2, g2, be2, sc2, sh2, 128);

    // BN2+ReLU + max over K, transposed store
    maxpool_kernel<<<(BATCH*SPTS*128)/256, 256, 0, stream>>>(Y3, sc2, sh2, out_pts);
}
